// FasterRCNN_39109972197405
// MI455X (gfx1250) — matrix operand order repaired and checked by reading, not compiled, not run on hardware
//
#include <hip/hip_runtime.h>
#include <hip/hip_bf16.h>

// CDNA5 / gfx1250: wave32, WMMA 16x16x32 bf16 -> f32 accumulate.

typedef __attribute__((ext_vector_type(16))) __bf16 bf16x16;
typedef __attribute__((ext_vector_type(4)))  __bf16 bf16x4;
typedef __attribute__((ext_vector_type(8)))  float  f32x8;

#define M_BATCH 4096
#define K_IN    12544
#define MID_N   1024
#define N_CLS   91
#define N_BBOX  364
#define N_HEAD  455
#define N_HPAD  512

// ---------------------------------------------------------------------------
// f32 -> bf16 conversion, 4 elements per thread (all sizes divisible by 4)
// ---------------------------------------------------------------------------
__global__ __launch_bounds__(256) void k_cvt4(const float4* __restrict__ src,
                                              bf16x4* __restrict__ dst,
                                              size_t n4) {
  size_t i = (size_t)blockIdx.x * blockDim.x + threadIdx.x;
  if (i < n4) {
    float4 f = src[i];
    bf16x4 o = {(__bf16)f.x, (__bf16)f.y, (__bf16)f.z, (__bf16)f.w};
    dst[i] = o;
  }
}

// ---------------------------------------------------------------------------
// Pack cls+bbox weights into one zero-padded [512, K] bf16 matrix
// ---------------------------------------------------------------------------
__global__ __launch_bounds__(256) void k_pack_head_w(const float* __restrict__ wc,
                                                     const float* __restrict__ wb,
                                                     __bf16* __restrict__ dst,
                                                     int K) {
  size_t i = (size_t)blockIdx.x * blockDim.x + threadIdx.x;
  size_t total = (size_t)N_HPAD * K;
  if (i >= total) return;
  int row = (int)(i / (size_t)K);
  int col = (int)(i - (size_t)row * K);
  float v = 0.0f;
  if (row < N_CLS)        v = wc[(size_t)row * K + col];
  else if (row < N_HEAD)  v = wb[(size_t)(row - N_CLS) * K + col];
  dst[i] = (__bf16)v;
}

__global__ __launch_bounds__(256) void k_pack_head_b(const float* __restrict__ bc,
                                                     const float* __restrict__ bb,
                                                     float* __restrict__ dst) {
  int i = blockIdx.x * blockDim.x + threadIdx.x;
  if (i >= N_HPAD) return;
  float v = 0.0f;
  if (i < N_CLS)        v = bc[i];
  else if (i < N_HEAD)  v = bb[i - N_CLS];
  dst[i] = v;
}

// ---------------------------------------------------------------------------
// bf16 WMMA GEMM:  C[M,N] = epi( A[M,K] * W[N,K]^T + bias )
//   Each wave owns a 32(M) x 64(N) tile = 2x4 grid of 16x16 WMMA accumulators.
//   A fragment (16x32 bf16, ISA 7.12.2): lane = hl*16+l16 holds row m0+l16,
//     K chunks [hl*8, hl*8+8) and [16+hl*8, 16+hl*8+8)  -> two b128 loads.
//   B fragment (32x16): lane holds col n0+l16, K chunks [hl*8, hl*8+8) and
//     [16+hl*8, 16+hl*8+8) of the W row (W stored [N,K] row-major).
//   EPI==0: ReLU, store bf16 [M,Npad]   (feeds next GEMM)
//   EPI==1: split f32 store into cls [M,91] and bbox [M,364]
// ---------------------------------------------------------------------------
template <int EPI>
__global__ __launch_bounds__(256) void gemm_wmma_bf16(
    const __bf16* __restrict__ A,
    const __bf16* __restrict__ W,
    const float*  __restrict__ bias,
    __bf16* __restrict__ Cb,
    float*  __restrict__ Fc,
    float*  __restrict__ Fb,
    int M, int Npad, int K) {
  const int wv   = (int)(threadIdx.x >> 5);
  const int lane = (int)(threadIdx.x & 31);
  const int gw   = blockIdx.x * (blockDim.x >> 5) + wv;
  const int nQuads = Npad >> 6;
  const int mT = gw / nQuads;
  const int nQ = gw - mT * nQuads;
  const int m0 = mT << 5;                 // 32 rows per wave
  if (m0 >= M) return;                    // wave-uniform: EXEC stays all-1s
  const int n0 = nQ << 6;                 // 64 cols per wave
  const int hl  = lane >> 4;
  const int l16 = lane & 15;

  const size_t arow0 = (size_t)(m0 + l16) * (size_t)K;
  const size_t arow1 = (size_t)(m0 + 16 + l16) * (size_t)K;
  size_t brow[4];
#pragma unroll
  for (int c = 0; c < 4; ++c) brow[c] = (size_t)(n0 + c * 16 + l16) * (size_t)K;

  f32x8 acc[2][4];
#pragma unroll
  for (int r = 0; r < 2; ++r)
#pragma unroll
    for (int c = 0; c < 4; ++c) acc[r][c] = (f32x8){0, 0, 0, 0, 0, 0, 0, 0};

  union Frag { bf16x16 v; uint4 q[2]; };

  for (int k = 0; k < K; k += 32) {
    Frag a0, a1;
    a0.q[0] = *(const uint4*)(A + arow0 + k + hl * 8);
    a0.q[1] = *(const uint4*)(A + arow0 + k + 16 + hl * 8);
    a1.q[0] = *(const uint4*)(A + arow1 + k + hl * 8);
    a1.q[1] = *(const uint4*)(A + arow1 + k + 16 + hl * 8);
#pragma unroll
    for (int c = 0; c < 4; ++c) {
      Frag b;
      const __bf16* wp = W + brow[c] + k + hl * 8;
      b.q[0] = *(const uint4*)(wp);
      b.q[1] = *(const uint4*)(wp + 16);
      acc[0][c] = __builtin_amdgcn_wmma_f32_16x16x32_bf16(
          false, a0.v, false, b.v, (short)0, acc[0][c], false, false);
      acc[1][c] = __builtin_amdgcn_wmma_f32_16x16x32_bf16(
          false, a1.v, false, b.v, (short)0, acc[1][c], false, false);
    }
  }

  // Epilogue. D layout (ISA 7.12.2): VGPR j, lane hl*16+l16 -> M = j + 8*hl,
  // N = l16 within each 16x16 tile.
#pragma unroll
  for (int c = 0; c < 4; ++c) {
    const int n = n0 + c * 16 + l16;
    const float bv = bias[n];
#pragma unroll
    for (int r = 0; r < 2; ++r) {
#pragma unroll
      for (int j = 0; j < 8; ++j) {
        const int m = m0 + r * 16 + j + 8 * hl;
        float v = acc[r][c][j] + bv;
        if (EPI == 0) {
          v = fmaxf(v, 0.0f);
          Cb[(size_t)m * Npad + n] = (__bf16)v;
        } else {
          if (n < N_CLS)
            Fc[(size_t)m * N_CLS + n] = v;
          else if (n < N_HEAD)
            Fb[(size_t)m * N_BBOX + (n - N_CLS)] = v;
        }
      }
    }
  }
}

// ---------------------------------------------------------------------------
extern "C" void kernel_launch(void* const* d_in, const int* in_sizes, int n_in,
                              void* d_out, int out_size, void* d_ws, size_t ws_size,
                              hipStream_t stream) {
  (void)in_sizes; (void)n_in; (void)out_size; (void)ws_size;
  const float* x  = (const float*)d_in[0];
  const float* w1 = (const float*)d_in[1];
  const float* b1 = (const float*)d_in[2];
  const float* w2 = (const float*)d_in[3];
  const float* b2 = (const float*)d_in[4];
  const float* wc = (const float*)d_in[5];
  const float* bc = (const float*)d_in[6];
  const float* wb = (const float*)d_in[7];
  const float* bb = (const float*)d_in[8];
  float* out = (float*)d_out;

  // Workspace carve-out (256B aligned chunks), ~148 MB total.
  char* p = (char*)d_ws;
  auto take = [&](size_t bytes) -> char* {
    char* q = p;
    p += (bytes + 255) & ~(size_t)255;
    return q;
  };
  __bf16* xb  = (__bf16*)take((size_t)M_BATCH * K_IN * 2);
  __bf16* w1b = (__bf16*)take((size_t)MID_N * K_IN * 2);
  __bf16* w2b = (__bf16*)take((size_t)MID_N * MID_N * 2);
  __bf16* whb = (__bf16*)take((size_t)N_HPAD * MID_N * 2);
  float*  bhb = (float*) take((size_t)N_HPAD * 4);
  __bf16* h1  = (__bf16*)take((size_t)M_BATCH * MID_N * 2);
  __bf16* h2  = (__bf16*)take((size_t)M_BATCH * MID_N * 2);

  // 1) f32 -> bf16 conversions (x lands in L2-resident bf16 form: 103 MB < 192 MB L2)
  {
    size_t n4 = (size_t)M_BATCH * K_IN / 4;
    k_cvt4<<<(unsigned)((n4 + 255) / 256), 256, 0, stream>>>((const float4*)x, (bf16x4*)xb, n4);
  }
  {
    size_t n4 = (size_t)MID_N * K_IN / 4;
    k_cvt4<<<(unsigned)((n4 + 255) / 256), 256, 0, stream>>>((const float4*)w1, (bf16x4*)w1b, n4);
  }
  {
    size_t n4 = (size_t)MID_N * MID_N / 4;
    k_cvt4<<<(unsigned)((n4 + 255) / 256), 256, 0, stream>>>((const float4*)w2, (bf16x4*)w2b, n4);
  }
  {
    size_t n = (size_t)N_HPAD * MID_N;
    k_pack_head_w<<<(unsigned)((n + 255) / 256), 256, 0, stream>>>(wc, wb, whb, MID_N);
  }
  k_pack_head_b<<<2, 256, 0, stream>>>(bc, bb, bhb);

  // 2) FC1: h1 = relu(xb @ w1^T + b1)   [4096 x 1024], K = 12544
  {
    int waves = (M_BATCH / 32) * (MID_N / 64);   // 2048
    gemm_wmma_bf16<0><<<waves / 8, 256, 0, stream>>>(
        xb, w1b, b1, h1, nullptr, nullptr, M_BATCH, MID_N, K_IN);
  }
  // 3) FC2: h2 = relu(h1 @ w2^T + b2)   [4096 x 1024], K = 1024
  {
    int waves = (M_BATCH / 32) * (MID_N / 64);   // 2048
    gemm_wmma_bf16<0><<<waves / 8, 256, 0, stream>>>(
        h1, w2b, b2, h2, nullptr, nullptr, M_BATCH, MID_N, MID_N);
  }
  // 4) Fused heads: [score | bbox] = h2 @ [w_cls; w_bbox]^T + bias, N padded to 512
  {
    int waves = (M_BATCH / 32) * (N_HPAD / 64);  // 1024
    gemm_wmma_bf16<1><<<waves / 8, 256, 0, stream>>>(
        h2, whb, bhb, nullptr, out, out + (size_t)M_BATCH * N_CLS,
        M_BATCH, N_HPAD, MID_N);
  }
}
